// KG_EdgeAtt_new_39419209842887
// MI455X (gfx1250) — compile-verified, matrix operand order checked by reading, not run on hardware
//
#include <hip/hip_runtime.h>

#define DEV __device__ __forceinline__

typedef __attribute__((ext_vector_type(16))) __bf16 v16bf;
typedef __attribute__((ext_vector_type(8)))  float  v8f;

constexpr int B_ = 32, L_ = 110, G_ = 512, N_ = 40, D_ = 300;
constexpr int LP = 112;      // padded L (multiple of 16)
constexpr int DP = 320;      // padded D (multiple of 32)
constexpr float EPSF  = 1e-8f;
constexpr float CLIPF = 1.0f - 1e-6f;
constexpr float PI_F  = 3.14159265358979323846f;

// ---- workspace layout (bytes) ----
constexpr size_t WSEM_OFF  = 0;                                   // Wsem bf16  [s][t] 512x512
constexpr size_t WCONT_OFF = WSEM_OFF  + (size_t)G_ * G_ * 2;     // WconT bf16 [t][s] 320x320 (zero padded)
constexpr size_t NFBF_OFF  = WCONT_OFF + (size_t)DP * DP * 2;     // nf bf16    [b][l][g] 32x112x512
constexpr size_t ATT_OFF   = NFBF_OFF  + (size_t)B_ * LP * G_ * 2;// att_sem bf16 [b][l][g]
constexpr size_t NFN_OFF   = ATT_OFF   + (size_t)B_ * LP * G_ * 2;// nf row norms f32 [b][l]
constexpr size_t NASQ_OFF  = NFN_OFF   + (size_t)B_ * LP * 4;     // att_sem row sq-norms f32 [b][l]

DEV unsigned short f2bf(float f) {
    unsigned int x = __float_as_uint(f);
    x += 0x7FFFu + ((x >> 16) & 1u);          // round-to-nearest-even
    return (unsigned short)(x >> 16);
}
DEV float bf2f(unsigned short h) { return __uint_as_float(((unsigned int)h) << 16); }

union FragU { unsigned int u[8]; v16bf v; };

// Load a 16x32 bf16 WMMA operand fragment.  All our operand sources are stored
// [out-dim][K] row-major (K-stride 1), so A and B use the identical gather:
//   row = lane%16, K pairs per VGPR v: base {0,2,4,6,16,18,20,22}[v] + (lane>=16 ? 8 : 0)
// (compiler merges these into two b128 loads)
DEV v16bf load_frag(const unsigned short* src, int row, int stride, int kk, int lane) {
    FragU f;
    const unsigned short* p = src + (size_t)row * stride + kk + ((lane >> 4) ? 8 : 0);
    f.u[0] = *(const unsigned int*)(p + 0);
    f.u[1] = *(const unsigned int*)(p + 2);
    f.u[2] = *(const unsigned int*)(p + 4);
    f.u[3] = *(const unsigned int*)(p + 6);
    f.u[4] = *(const unsigned int*)(p + 16);
    f.u[5] = *(const unsigned int*)(p + 18);
    f.u[6] = *(const unsigned int*)(p + 20);
    f.u[7] = *(const unsigned int*)(p + 22);
    return f.v;
}

DEV v8f zero8() {
    v8f a;
    #pragma unroll
    for (int i = 0; i < 8; ++i) a[i] = 0.f;
    return a;
}

DEV v8f wmma_bf16(v16bf a, v16bf b, v8f c) {
    return __builtin_amdgcn_wmma_f32_16x16x32_bf16(false, a, false, b, (short)0, c, false, false);
}

// ---- CDNA5 async global->LDS copy (ASYNCcnt-tracked), GV addressing mode ----
DEV void ld_async_b128(void* lds_ptr, const void* gptr) {
    unsigned lds_off = (unsigned)(size_t)lds_ptr;            // low 32 bits of generic = LDS offset
    unsigned long long ga = (unsigned long long)(size_t)gptr;
    asm volatile("global_load_async_to_lds_b128 %0, %1, off"
                 :: "v"(lds_off), "v"(ga) : "memory");
}
DEV void wait_async0() { asm volatile("s_wait_asynccnt 0x0" ::: "memory"); }

// ---------------- prep: weights -> bf16 ----------------
__global__ void k_prep_w(const float* __restrict__ wsem, const float* __restrict__ wcon,
                         unsigned short* __restrict__ wsem_bf, unsigned short* __restrict__ wconT_bf) {
    int idx = blockIdx.x * 256 + threadIdx.x;
    if (idx < G_ * G_) wsem_bf[idx] = f2bf(wsem[idx]);
    if (idx < DP * DP) {
        int t = idx / DP, s = idx % DP;
        float v = (s < D_ && t < D_) ? wcon[s * D_ + t] : 0.f;
        wconT_bf[idx] = f2bf(v);
    }
}

// ---------------- prep: node features -> bf16 + row norms ----------------
__global__ void k_prep_nf(const float* __restrict__ nf, unsigned short* __restrict__ nf_bf,
                          float* __restrict__ nf_norm) {
    int row = blockIdx.x;                 // b*LP + l
    int b = row / LP, l = row % LP;
    int tid = threadIdx.x;                // 128 threads
    __shared__ float red[128];
    float part = 0.f;
    for (int g = tid; g < G_; g += 128) {
        float v = (l < L_) ? nf[((size_t)b * L_ + l) * G_ + g] : 0.f;
        nf_bf[(size_t)row * G_ + g] = f2bf(v);
        part += v * v;
    }
    red[tid] = part;
    __syncthreads();
    for (int s = 64; s > 0; s >>= 1) { if (tid < s) red[tid] += red[tid + s]; __syncthreads(); }
    if (tid == 0) nf_norm[row] = sqrtf(red[0]);
}

// ---------------- S1: att_sem = nf @ Wsem^T (WMMA), + na^2 ----------------
__global__ void __launch_bounds__(256) k_att_sem(const unsigned short* __restrict__ nf_bf,
                                                 const unsigned short* __restrict__ wsem_bf,
                                                 unsigned short* __restrict__ att_bf,
                                                 float* __restrict__ na_sq) {
    int b = blockIdx.x / 7, mt = blockIdx.x % 7;
    __shared__ unsigned short Alds[16 * G_];        // 16 KB A tile
    // Async-stage the A tile: 16*512*2 = 16384 B = 4 passes of 256 lanes * 16 B
    {
        const unsigned char* src = (const unsigned char*)(nf_bf + ((size_t)b * LP + mt * 16) * G_);
        unsigned char* dst = (unsigned char*)Alds;
        #pragma unroll
        for (int it = 0; it < 4; ++it) {
            int off = it * 4096 + threadIdx.x * 16;
            ld_async_b128(dst + off, src + off);
        }
        wait_async0();
    }
    __syncthreads();

    int wave = threadIdx.x >> 5, lane = threadIdx.x & 31;
    int hf = lane >> 4, c16 = lane & 15;
    for (int nt = wave; nt < G_ / 16; nt += 8) {
        int brow = nt * 16 + c16;
        v8f acc = zero8();
        v16bf bcur = load_frag(wsem_bf, brow, G_, 0, lane);
        for (int kk = 0; kk < G_; kk += 32) {
            v16bf bnext = bcur;
            if (kk + 32 < G_) bnext = load_frag(wsem_bf, brow, G_, kk + 32, lane);
            v16bf a = load_frag(Alds, c16, G_, kk, lane);
            acc = wmma_bf16(a, bcur, acc);
            bcur = bnext;
        }
        int scol = nt * 16 + c16;
        #pragma unroll
        for (int v = 0; v < 8; ++v) {
            int l = mt * 16 + v + hf * 8;
            float c = acc[v];
            att_bf[((size_t)b * LP + l) * G_ + scol] = f2bf(c);
            float s = c * c;
            s += __shfl_xor(s, 1, 32);
            s += __shfl_xor(s, 2, 32);
            s += __shfl_xor(s, 4, 32);
            s += __shfl_xor(s, 8, 32);
            if (c16 == 0) atomicAdd(&na_sq[b * LP + l], s);
        }
    }
}

// ---------------- S2: banded sem scores + softmax -> out = 0.5*alphas_sem ----------------
__global__ void k_sem_softmax(const unsigned short* __restrict__ nf_bf,
                              const unsigned short* __restrict__ att_bf,
                              const float* __restrict__ nf_norm,
                              const float* __restrict__ na_sq,
                              const int* __restrict__ text_len,
                              float* __restrict__ out) {
    int b = blockIdx.x / 7, jt = blockIdx.x % 7;
    int lane = threadIdx.x;               // 32 threads = 1 wave
    int hf = lane >> 4, c16 = lane & 15;
    __shared__ float sc[16][48];
    int cur = text_len[b];
    const unsigned short* nfB = nf_bf + (size_t)b * LP * G_;
    const unsigned short* atB = att_bf + (size_t)b * LP * G_;

    for (int kti = 0; kti < 3; ++kti) {
        int kt = jt - 1 + kti;
        bool valid = (kt >= 0) && (kt < 7);
        v8f acc = zero8();
        if (valid) {
            int arow = jt * 16 + c16, brow = kt * 16 + c16;
            v16bf acur = load_frag(nfB, arow, G_, 0, lane);
            v16bf bcur = load_frag(atB, brow, G_, 0, lane);
            for (int kk = 0; kk < G_; kk += 32) {
                v16bf anext = acur, bnext = bcur;
                if (kk + 32 < G_) {
                    anext = load_frag(nfB, arow, G_, kk + 32, lane);
                    bnext = load_frag(atB, brow, G_, kk + 32, lane);
                }
                acc = wmma_bf16(acur, bcur, acc);
                acur = anext; bcur = bnext;
            }
        }
        int kg = kt * 16 + c16;
        #pragma unroll
        for (int v = 0; v < 8; ++v) {
            int j = jt * 16 + v + hf * 8;
            float scv = -__builtin_inff();
            if (valid && j < L_ && kg >= 0 && kg < L_ &&
                kg >= j - 10 && kg <= j + 10 && j < cur && kg <= cur - 1) {
                float dn = fmaxf(nf_norm[b * LP + j], EPSF) *
                           fmaxf(sqrtf(na_sq[b * LP + kg]), EPSF);
                float cs = acc[v] / dn;
                cs = fminf(fmaxf(cs, -CLIPF), CLIPF);
                scv = 1.0f - acosf(cs) * (1.0f / PI_F);
            }
            sc[v + hf * 8][kti * 16 + c16] = scv;
        }
    }
    __syncthreads();
    if (lane < 16) {
        int j = jt * 16 + lane;
        if (j < L_ && j < cur) {
            float m = -__builtin_inff();
            for (int c = 0; c < 48; ++c) m = fmaxf(m, sc[lane][c]);
            if (m > -1e30f) {
                float sum = 0.f;
                for (int c = 0; c < 48; ++c) {
                    float e = expf(sc[lane][c] - m);   // exp(-inf)=0 for masked
                    sc[lane][c] = e;
                    sum += e;
                }
                float inv = 0.5f / fmaxf(sum, EPSF);
                for (int c = 0; c < 48; ++c) {
                    int kg = (jt - 1) * 16 + c;
                    if (kg >= 0 && kg < L_)
                        out[((size_t)b * L_ + j) * L_ + kg] = sc[lane][c] * inv;
                }
            }
        }
    }
}

// ---------------- Con path: per (b,n) -> out += 0.5*alphas_con ----------------
__global__ void __launch_bounds__(256) k_con(const float* __restrict__ know,
                                             const unsigned short* __restrict__ wconT,
                                             const float* __restrict__ anew,
                                             const int* __restrict__ text_len,
                                             float* __restrict__ out) {
    int b = blockIdx.x / N_, n = blockIdx.x % N_;
    __shared__ unsigned short Ks[LP * DP];   // knowledge slice bf16, [l][d]
    __shared__ unsigned short Af[LP * DP];   // aff_feat bf16,        [l][t]
    __shared__ float nk_sq[LP];
    __shared__ float nf_sq[LP];
    int tid = threadIdx.x;

    for (int idx = tid; idx < LP * DP; idx += 256) {
        int l = idx / DP, d = idx % DP;
        float v = 0.f;
        if (l < L_ && d < D_) v = know[(((size_t)b * L_ + l) * N_ + n) * D_ + d];
        Ks[idx] = f2bf(v);
    }
    __syncthreads();
    if (tid < LP) {
        float s = 0.f;
        for (int d = 0; d < D_; ++d) { float v = bf2f(Ks[tid * DP + d]); s += v * v; }
        nk_sq[tid] = s;
        nf_sq[tid] = 0.f;
    }
    __syncthreads();

    int wave = tid >> 5, lane = tid & 31, hf = lane >> 4, c16 = lane & 15;

    // GEMM-1: att_con = Ks @ Wcon, scale by anew_aff, store bf16 to Af, accumulate nf^2
    for (int job = wave; job < 7 * (DP / 16); job += 8) {
        int mt = job / (DP / 16), nt = job % (DP / 16);
        int arow = mt * 16 + c16, brow = nt * 16 + c16;
        v8f acc = zero8();
        v16bf bcur = load_frag(wconT, brow, DP, 0, lane);      // global: double-buffer
        for (int kk = 0; kk < DP; kk += 32) {
            v16bf bnext = bcur;
            if (kk + 32 < DP) bnext = load_frag(wconT, brow, DP, kk + 32, lane);
            v16bf a = load_frag(Ks, arow, DP, kk, lane);
            acc = wmma_bf16(a, bcur, acc);
            bcur = bnext;
        }
        int t = nt * 16 + c16;
        #pragma unroll
        for (int v = 0; v < 8; ++v) {
            int l = mt * 16 + v + hf * 8;
            float scl = 0.f;
            if (l < L_) {
                float a0 = anew[((size_t)b * L_ + l) * N_ + n];
                scl = (sqrtf((a0 - 0.5f) * (a0 - 0.5f) + 0.25f * a0 * a0) - 0.06467f) *
                      (1.0f / 0.607468f);
            }
            float av = acc[v] * scl;
            Af[l * DP + t] = f2bf(av);
            float s = av * av;
            s += __shfl_xor(s, 1, 32);
            s += __shfl_xor(s, 2, 32);
            s += __shfl_xor(s, 4, 32);
            s += __shfl_xor(s, 8, 32);
            if (c16 == 0) atomicAdd(&nf_sq[l], s);
        }
    }
    __syncthreads();

    int cur = text_len[b];
    // GEMM-2: banded num_c = Ks[j] . Af[k];  accumulate 5*|cos| into out
    for (int job = wave; job < 21; job += 8) {
        int jt = job / 3, kt = jt - 1 + job % 3;
        if (kt < 0 || kt > 6) continue;
        v8f acc = zero8();
        for (int kk = 0; kk < DP; kk += 32) {
            v16bf a  = load_frag(Ks, jt * 16 + c16, DP, kk, lane);
            v16bf bb = load_frag(Af, kt * 16 + c16, DP, kk, lane);
            acc = wmma_bf16(a, bb, acc);
        }
        int k = kt * 16 + c16;
        #pragma unroll
        for (int v = 0; v < 8; ++v) {
            int j = jt * 16 + v + hf * 8;
            if (j < L_ && k < L_ && k >= j - 10 && k <= j + 10 && j < cur && k <= cur - 1) {
                float dn = fmaxf(sqrtf(nf_sq[k]), EPSF) * fmaxf(sqrtf(nk_sq[j]), EPSF);
                atomicAdd(&out[((size_t)b * L_ + j) * L_ + k], 5.0f * fabsf(acc[v] / dn));
            }
        }
    }
}

extern "C" void kernel_launch(void* const* d_in, const int* in_sizes, int n_in,
                              void* d_out, int out_size, void* d_ws, size_t ws_size,
                              hipStream_t stream) {
    (void)in_sizes; (void)n_in; (void)ws_size;
    const float* nf   = (const float*)d_in[0];
    const float* know = (const float*)d_in[1];
    const float* anew = (const float*)d_in[2];
    const float* wsem = (const float*)d_in[3];
    const float* wcon = (const float*)d_in[4];
    const int*   tlen = (const int*)d_in[5];
    float* out = (float*)d_out;

    unsigned char* ws = (unsigned char*)d_ws;
    unsigned short* wsem_bf  = (unsigned short*)(ws + WSEM_OFF);
    unsigned short* wconT_bf = (unsigned short*)(ws + WCONT_OFF);
    unsigned short* nf_bf    = (unsigned short*)(ws + NFBF_OFF);
    unsigned short* att_bf   = (unsigned short*)(ws + ATT_OFF);
    float*          nf_norm  = (float*)(ws + NFN_OFF);
    float*          na_sq    = (float*)(ws + NASQ_OFF);

    (void)hipMemsetAsync(d_out, 0, (size_t)out_size * sizeof(float), stream);
    (void)hipMemsetAsync(na_sq, 0, (size_t)B_ * LP * sizeof(float), stream);

    k_prep_w<<<(G_ * G_ + 255) / 256, 256, 0, stream>>>(wsem, wcon, wsem_bf, wconT_bf);
    k_prep_nf<<<B_ * LP, 128, 0, stream>>>(nf, nf_bf, nf_norm);
    k_att_sem<<<B_ * 7, 256, 0, stream>>>(nf_bf, wsem_bf, att_bf, na_sq);
    k_sem_softmax<<<B_ * 7, 32, 0, stream>>>(nf_bf, att_bf, nf_norm, na_sq, tlen, out);
    k_con<<<B_ * N_, 256, 0, stream>>>(know, wconT_bf, anew, tlen, out);
}
